// VectorQuantizer_16569983828148
// MI455X (gfx1250) — matrix operand order repaired and checked by reading, not compiled, not run on hardware
//
#include <hip/hip_runtime.h>

typedef __attribute__((ext_vector_type(16))) __bf16 v16bf;
typedef __attribute__((ext_vector_type(8)))  float  v8f;
typedef __attribute__((ext_vector_type(4)))  unsigned int v4u;

#define K_CODES      512
#define DDIM         64
#define ROW_PAD      72          // 64 + 8 shorts padding -> 144B row stride, conflict-free 16B LDS reads
#define ROWS_PER_BLK 256
#define THREADS      256
#define N_TOTAL      8388608.0f  // B*D*H*W elements for the mean

// d_ws layout (bytes):
//   [0,        2048)  : 512 fp32 per-block loss partials
//   [2048,    75776)  : padded bf16 codebook, K_CODES * ROW_PAD shorts (16B aligned)
//   [75776,   77824)  : 512 fp32 ||e_k||^2
#define WS_EMBBF_OFF 2048
#define WS_E2_OFF    75776

union FragU { v16bf b; v4u u[2]; };

__device__ __forceinline__ unsigned short f2bf(float f) {
  unsigned int u = __float_as_uint(f);
  u += 0x7FFFu + ((u >> 16) & 1u);   // round-to-nearest-even
  return (unsigned short)(u >> 16);
}

__device__ __forceinline__ float bf2f(unsigned short h) {
  return __uint_as_float(((unsigned int)h) << 16);
}

// ---- one-shot prep: padded bf16 codebook + ||e||^2 into d_ws ----
__global__ __launch_bounds__(256)
void vq_prep(const float* __restrict__ emb,
             unsigned short* __restrict__ emb_bf,
             float* __restrict__ e2) {
  int c = blockIdx.x * 256 + threadIdx.x;          // 0..511
  const float* row = emb + c * DDIM;
  float s = 0.f;
  #pragma unroll
  for (int d = 0; d < DDIM; ++d) {
    unsigned short h = f2bf(row[d]);
    emb_bf[c * ROW_PAD + d] = h;
    float v = bf2f(h);
    s += v * v;
  }
  e2[c] = s;
}

__global__ __launch_bounds__(THREADS)
void vq_kernel(const float* __restrict__ latents,
               const float* __restrict__ emb,
               const unsigned short* __restrict__ emb_bf,
               const float* __restrict__ e2g,
               float* __restrict__ out,
               float* __restrict__ partial) {
  __shared__ __align__(16) unsigned short emb_lds[K_CODES * ROW_PAD];      // 73728 B
  __shared__ __align__(16) unsigned short x_lds[ROWS_PER_BLK * ROW_PAD];   // 36864 B
  __shared__ __align__(16) float e2_lds[K_CODES];                          // 2048 B
  __shared__ int   idx_lds[ROWS_PER_BLK];                                  // 1024 B

  const int tid = threadIdx.x;
  const int blk = blockIdx.x;
  const int b   = blk >> 4;                 // 16 blocks per image (4096 pixels / 256)
  const int p   = (blk & 15) * ROWS_PER_BLK;
  const long img = (long)b * (DDIM * 4096);

  // ---- async block-copy of the prepped bf16 codebook + e2 into LDS (ASYNCcnt path) ----
  {
    unsigned emb_base = (unsigned)(uintptr_t)(&emb_lds[0]);
    #pragma unroll
    for (int j = 0; j < (K_CODES * ROW_PAD * 2) / (THREADS * 16); ++j) {  // 18 iters
      unsigned off = (unsigned)(j * THREADS + tid) * 16u;
      asm volatile("global_load_async_to_lds_b128 %0, %1, %2"
                   :: "v"(emb_base + off), "v"(off), "s"(emb_bf)
                   : "memory");
    }
    if (tid < (K_CODES * 4) / 16) {                                        // 128 lanes
      unsigned e2_base = (unsigned)(uintptr_t)(&e2_lds[0]);
      unsigned off = (unsigned)tid * 16u;
      asm volatile("global_load_async_to_lds_b128 %0, %1, %2"
                   :: "v"(e2_base + off), "v"(off), "s"(e2g)
                   : "memory");
    }
  }

  // ---- stage bf16 x-tile into LDS (VALU convert overlaps the async copies) ----
  #pragma unroll
  for (int j = 0; j < (ROWS_PER_BLK * DDIM) / THREADS; ++j) {
    int flat = j * THREADS + tid;
    int d = flat >> 8;
    int r = flat & 255;
    x_lds[r * ROW_PAD + d] = f2bf(latents[img + (long)d * 4096 + p + r]);
  }

  asm volatile("s_wait_asynccnt 0" ::: "memory");
  __syncthreads();

  const int lane = tid & 31;
  const int wave = tid >> 5;
  const int m  = lane & 15;
  const int kh = lane >> 4;
  const int rbase = wave * 32;              // each wave owns 32 rows (two A tiles)

  // A fragments per ISA 16-bit A 16x32 layout: V0-3 <- K[8kh..8kh+7], V4-7 <- K[16+8kh..+7]
  FragU a00, a01, a10, a11;
  {
    const unsigned short* xr0 = &x_lds[(rbase + m) * ROW_PAD];
    const unsigned short* xr1 = &x_lds[(rbase + 16 + m) * ROW_PAD];
    a00.u[0] = *(const v4u*)(xr0 +  8 * kh);
    a00.u[1] = *(const v4u*)(xr0 + 16 + 8 * kh);
    a01.u[0] = *(const v4u*)(xr0 + 32 + 8 * kh);
    a01.u[1] = *(const v4u*)(xr0 + 48 + 8 * kh);
    a10.u[0] = *(const v4u*)(xr1 +  8 * kh);
    a10.u[1] = *(const v4u*)(xr1 + 16 + 8 * kh);
    a11.u[0] = *(const v4u*)(xr1 + 32 + 8 * kh);
    a11.u[1] = *(const v4u*)(xr1 + 48 + 8 * kh);
  }

  float best0[8], best1[8];
  int   bidx0[8], bidx1[8];
  #pragma unroll
  for (int j = 0; j < 8; ++j) {
    best0[j] = 3.4e38f; bidx0[j] = 0;
    best1[j] = 3.4e38f; bidx1[j] = 0;
  }

  #pragma unroll 2
  for (int ct = 0; ct < K_CODES / 16; ++ct) {
    const int c = ct * 16 + m;
    const unsigned short* er = &emb_lds[c * ROW_PAD];
    // B fragment: lane holds K[8kh..8kh+7] then K[16+8kh..16+8kh+7] of column c (matches A)
    FragU b0, b1;
    b0.u[0] = *(const v4u*)(er + 8 * kh);
    b0.u[1] = *(const v4u*)(er + 16 + 8 * kh);
    b1.u[0] = *(const v4u*)(er + 32 + 8 * kh);
    b1.u[1] = *(const v4u*)(er + 48 + 8 * kh);

    // two independent accumulator chains reuse the same B fragments (4 WMMAs / tile)
    v8f z = {0.f, 0.f, 0.f, 0.f, 0.f, 0.f, 0.f, 0.f};
    v8f acc0 = __builtin_amdgcn_wmma_f32_16x16x32_bf16(false, a00.b, false, b0.b,
                                                       (short)0, z, false, false);
    v8f acc1 = __builtin_amdgcn_wmma_f32_16x16x32_bf16(false, a10.b, false, b0.b,
                                                       (short)0, z, false, false);
    acc0 = __builtin_amdgcn_wmma_f32_16x16x32_bf16(false, a01.b, false, b1.b,
                                                   (short)0, acc0, false, false);
    acc1 = __builtin_amdgcn_wmma_f32_16x16x32_bf16(false, a11.b, false, b1.b,
                                                   (short)0, acc1, false, false);

    const float e2c = e2_lds[c];
    #pragma unroll
    for (int j = 0; j < 8; ++j) {
      float s0 = e2c - 2.f * acc0[j];
      float s1 = e2c - 2.f * acc1[j];
      if (s0 < best0[j]) { best0[j] = s0; bidx0[j] = c; }  // c strictly increases -> first-min kept
      if (s1 < best1[j]) { best1[j] = s1; bidx1[j] = c; }
    }
  }

  // ---- 16-lane min+index reduction (rows j / j+8 live in lane halves) ----
  #pragma unroll
  for (int j = 0; j < 8; ++j) {
    float bs0 = best0[j]; int bi0 = bidx0[j];
    float bs1 = best1[j]; int bi1 = bidx1[j];
    #pragma unroll
    for (int mask = 8; mask >= 1; mask >>= 1) {
      float os0 = __shfl_xor(bs0, mask, 32);
      int   oi0 = __shfl_xor(bi0, mask, 32);
      float os1 = __shfl_xor(bs1, mask, 32);
      int   oi1 = __shfl_xor(bi1, mask, 32);
      if (os0 < bs0 || (os0 == bs0 && oi0 < bi0)) { bs0 = os0; bi0 = oi0; }
      if (os1 < bs1 || (os1 == bs1 && oi1 < bi1)) { bs1 = os1; bi1 = oi1; }
    }
    if (m == 0) {
      idx_lds[rbase + j + 8 * kh]      = bi0;
      idx_lds[rbase + 16 + j + 8 * kh] = bi1;
    }
  }
  __syncthreads();

  // ---- write q_st = x + (q - x) in [B,D,H,W] order, accumulate loss ----
  float lsum = 0.f;
  #pragma unroll
  for (int j = 0; j < (ROWS_PER_BLK * DDIM) / THREADS; ++j) {
    int flat = j * THREADS + tid;
    int d  = flat >> 8;
    int rr = flat & 255;
    long gi = img + (long)d * 4096 + p + rr;
    float q = emb[idx_lds[rr] * DDIM + d];   // fp32 codebook, L2-resident
    float x = latents[gi];
    out[gi] = x + (q - x);
    float dd = q - x;
    lsum += dd * dd;
  }
  #pragma unroll
  for (int off = 16; off >= 1; off >>= 1) lsum += __shfl_xor(lsum, off, 32);
  if (lane == 0) e2_lds[wave] = lsum;        // reuse e2 LDS as scratch
  __syncthreads();
  if (tid == 0) {
    float t = 0.f;
    #pragma unroll
    for (int w = 0; w < THREADS / 32; ++w) t += e2_lds[w];
    partial[blk] = t;
  }
}

__global__ __launch_bounds__(256)
void vq_loss_reduce(const float* __restrict__ partial, float* __restrict__ out_scalar) {
  __shared__ float s_lds[8];
  int tid = threadIdx.x;
  float s = partial[tid] + partial[tid + 256];
  #pragma unroll
  for (int off = 16; off >= 1; off >>= 1) s += __shfl_xor(s, off, 32);
  if ((tid & 31) == 0) s_lds[tid >> 5] = s;
  __syncthreads();
  if (tid == 0) {
    float t = 0.f;
    #pragma unroll
    for (int w = 0; w < 8; ++w) t += s_lds[w];
    // vq_loss = commitment*0.25 + embedding, both == mean((q-x)^2)
    out_scalar[0] = t * (1.25f / N_TOTAL);
  }
}

extern "C" void kernel_launch(void* const* d_in, const int* in_sizes, int n_in,
                              void* d_out, int out_size, void* d_ws, size_t ws_size,
                              hipStream_t stream) {
  const float* latents = (const float*)d_in[0];   // [32,64,64,64] fp32
  const float* emb     = (const float*)d_in[1];   // [512,64] fp32
  float* out = (float*)d_out;                     // 8388608 q_st + 1 loss

  float*          partial = (float*)d_ws;                                    // 512 fp32
  unsigned short* emb_bf  = (unsigned short*)((char*)d_ws + WS_EMBBF_OFF);   // padded bf16 codebook
  float*          e2      = (float*)((char*)d_ws + WS_E2_OFF);               // 512 fp32

  vq_prep<<<dim3(2), dim3(256), 0, stream>>>(emb, emb_bf, e2);
  vq_kernel<<<dim3(512), dim3(THREADS), 0, stream>>>(latents, emb, emb_bf, e2, out, partial);
  vq_loss_reduce<<<dim3(1), dim3(256), 0, stream>>>(partial, out + (out_size - 1));
}
